// VectorQuantizerEMA_10668698763636
// MI455X (gfx1250) — compile-verified, matrix operand order checked by reading, not run on hardware
//
#include <hip/hip_runtime.h>

typedef __bf16 v16bf __attribute__((ext_vector_type(16)));
typedef float  v8f   __attribute__((ext_vector_type(8)));
typedef float  v4f   __attribute__((ext_vector_type(4)));

#define DDIM   256
#define KCODES 1024
#define XPAD   260          // 260 % 64 == 4 -> conflict-free column access
#define DECAYF 0.99f
#define ONE_MINUS_DECAYF 0.01f
#define EPSF   1e-5f

// K-dim index held by element e of a 16-bit WMMA operand fragment, for lane half h
// (CDNA5 ISA 7.12.2, 16-bit A-Matrix 16x32 table; B assumed symmetric with lane=column)
__device__ __forceinline__ int kmap(int e, int half) {
    return (e & 7) + 8 * (2 * (e >> 3) + half);
}

// ---------------------------------------------------------------------------
// Kernel 1: decompose embedding into bf16 hi/lo WMMA B-fragments + ||e||^2
// grid: 512 blocks (64 code-tiles x 8 K-chunks) x 32 threads (one wave)
// ---------------------------------------------------------------------------
__global__ __launch_bounds__(32) void vq_prepack(
    const float* __restrict__ embedding,   // [K, D]
    v16bf* __restrict__ ehi,               // [64*8*32] fragments
    v16bf* __restrict__ elo,
    float* __restrict__ enorm)             // [K], pre-zeroed
{
    const int lane = threadIdx.x;
    const int half = lane >> 4;
    const int idx  = lane & 15;
    const int tile  = blockIdx.x >> 3;     // 0..63  (16 codes each)
    const int chunk = blockIdx.x & 7;      // 0..7   (32 d each)
    const int code  = tile * 16 + idx;

    v16bf h, l;
    float ss = 0.0f;
#pragma unroll
    for (int e = 0; e < 16; ++e) {
        int d = chunk * 32 + kmap(e, half);
        float x = embedding[code * DDIM + d];
        __bf16 hh = (__bf16)x;
        h[e] = hh;
        l[e] = (__bf16)(x - (float)hh);
        ss += x * x;
    }
    ehi[blockIdx.x * 32 + lane] = h;
    elo[blockIdx.x * 32 + lane] = l;

    ss += __shfl_xor(ss, 16, 32);          // combine the two halves per code
    if (half == 0) atomicAdd(&enorm[code], ss);
}

// ---------------------------------------------------------------------------
// Kernel 2: fused distance-WMMA + argmin + all big outputs
// grid: 4096 blocks (16 x-rows each) x 256 threads (8 waves)
// ---------------------------------------------------------------------------
__global__ __launch_bounds__(256) void vq_main(
    const float* __restrict__ inputs,      // [B,C,H,W] = [64,256,32,32]
    const float* __restrict__ embedding,   // [K, D]
    const v16bf* __restrict__ ehi,
    const v16bf* __restrict__ elo,
    const float* __restrict__ enorm,
    float* __restrict__ out,               // [B,C,H,W]
    float* __restrict__ encodings,         // [N, K]
    float* __restrict__ dw,                // [K, D] pre-zeroed, atomic accum
    float* __restrict__ counts,            // [K]    pre-zeroed
    float* __restrict__ loss_part)         // [4096] per-block partials
{
    __shared__ float xtile[16 * XPAD];
    __shared__ float qtile[16 * XPAD];
    __shared__ float redmin[8 * 16];
    __shared__ int   redarg[8 * 16];
    __shared__ int   idx16[16];
    __shared__ float lred[256];

    const int t   = threadIdx.x;
    const int blk = blockIdx.x;            // 0..4095
    const int b   = blk >> 6;              // image
    const int hw0 = (blk & 63) << 4;       // first of 16 consecutive hw
    const size_t ibase = (size_t)b * (DDIM * 1024);

    // ---- stage x-tile [16 hw][256 d] (NCHW -> coalesced: 16 consecutive hw) ----
#pragma unroll
    for (int i = 0; i < 16; ++i) {
        int d   = (t >> 4) + 16 * i;
        int hwi = t & 15;
        xtile[hwi * XPAD + d] = inputs[ibase + (size_t)d * 1024 + hw0 + hwi];
    }
    __syncthreads();

    const int wv   = t >> 5;
    const int lane = t & 31;
    const int half = lane >> 4;
    const int idx  = lane & 15;

    // ---- build A fragments (hi/lo) once per wave, kept in registers ----
    v16bf Ahi[8], Alo[8];
#pragma unroll
    for (int c = 0; c < 8; ++c) {
#pragma unroll
        for (int e = 0; e < 16; ++e) {
            int d = c * 32 + kmap(e, half);
            float x = xtile[idx * XPAD + d];
            __bf16 hh = (__bf16)x;
            Ahi[c][e] = hh;
            Alo[c][e] = (__bf16)(x - (float)hh);
        }
    }

    float best[8];
    int   bestn[8];
#pragma unroll
    for (int r = 0; r < 8; ++r) { best[r] = 3.0e38f; bestn[r] = 0; }

    // ---- 8 code-tiles per wave: S = x . e^T via bf16x3 WMMA, f32 accum ----
    for (int it = 0; it < 8; ++it) {
        const int tile = wv + (it << 3);   // 0..63
        v8f acc = {0.f, 0.f, 0.f, 0.f, 0.f, 0.f, 0.f, 0.f};
#pragma unroll
        for (int c = 0; c < 8; ++c) {
            const v16bf bh = ehi[(tile * 8 + c) * 32 + lane];
            const v16bf bl = elo[(tile * 8 + c) * 32 + lane];
            acc = __builtin_amdgcn_wmma_f32_16x16x32_bf16(false, Ahi[c], false, bh, (short)0, acc, false, false);
            acc = __builtin_amdgcn_wmma_f32_16x16x32_bf16(false, Alo[c], false, bh, (short)0, acc, false, false);
            acc = __builtin_amdgcn_wmma_f32_16x16x32_bf16(false, Ahi[c], false, bl, (short)0, acc, false, false);
        }
        const int n  = tile * 16 + idx;    // code index owned by this lane
        const float en = enorm[n];
#pragma unroll
        for (int r = 0; r < 8; ++r) {
            float dist = en - 2.0f * acc[r];   // ||x||^2 dropped (const per row)
            if (dist < best[r] || (dist == best[r] && n < bestn[r])) {
                best[r] = dist; bestn[r] = n;
            }
        }
    }

    // ---- argmin reduce across the 16 column-lanes of each half ----
#pragma unroll
    for (int off = 8; off >= 1; off >>= 1) {
#pragma unroll
        for (int r = 0; r < 8; ++r) {
            float ob = __shfl_xor(best[r], off, 32);
            int   on = __shfl_xor(bestn[r], off, 32);
            if (ob < best[r] || (ob == best[r] && on < bestn[r])) {
                best[r] = ob; bestn[r] = on;
            }
        }
    }
    if (idx == 0) {
#pragma unroll
        for (int r = 0; r < 8; ++r) {
            int m = r + 8 * half;          // row of the D-matrix this element holds
            redmin[wv * 16 + m] = best[r];
            redarg[wv * 16 + m] = bestn[r];
        }
    }
    __syncthreads();
    if (t < 16) {                          // reduce across the 8 waves
        float bm = redmin[t];
        int   bn = redarg[t];
#pragma unroll
        for (int w2 = 1; w2 < 8; ++w2) {
            float om = redmin[w2 * 16 + t];
            int   on = redarg[w2 * 16 + t];
            if (om < bm || (om == bm && on < bn)) { bm = om; bn = on; }
        }
        idx16[t] = bn;
        atomicAdd(&counts[bn], 1.0f);      // exact in f32 (<= 65536)
    }
    __syncthreads();

    // ---- pass A: quantize (embedding-friendly), loss, dw scatter ----
    float lsum = 0.0f;
    {
        const int m = t >> 4;
        const int j = t & 15;
        const int code = idx16[m];
        const float* erow  = embedding + (size_t)code * DDIM;
        float*       dwrow = dw        + (size_t)code * DDIM;
#pragma unroll
        for (int i = 0; i < 16; ++i) {
            int c = j + 16 * i;            // coalesced 64B per row-group
            float q  = erow[c];
            float x  = xtile[m * XPAD + c];
            float dq = q - x;
            lsum += dq * dq;
            qtile[m * XPAD + c] = x + dq;  // straight-through value
            atomicAdd(&dwrow[c], x);       // contiguous 64B atomic groups
        }
    }
    lred[t] = lsum;
    __syncthreads();
    for (int s = 128; s > 0; s >>= 1) {
        if (t < s) lred[t] += lred[t + s];
        __syncthreads();
    }
    if (t == 0) loss_part[blk] = lred[0];  // deterministic per-block slot

    // ---- pass B: out (NCHW, coalesced over hw) ----
#pragma unroll
    for (int i = 0; i < 16; ++i) {
        int c   = (t >> 4) + 16 * i;
        int hwi = t & 15;
        out[ibase + (size_t)c * 1024 + hw0 + hwi] = qtile[hwi * XPAD + c];
    }

    // ---- one-hot encodings tile: 16 rows x 1024, b128 streaming stores ----
    {
        const int m = t >> 4;
        const int j = t & 15;
        const int code = idx16[m];
        const size_t rowbase = (size_t)(blk * 16 + m) * KCODES;
#pragma unroll
        for (int i = 0; i < 16; ++i) {
            int k = j * 4 + 64 * i;        // consecutive lanes -> consecutive float4
            v4f v = {0.f, 0.f, 0.f, 0.f};
            if (code >= k && code < k + 4) v[code - k] = 1.0f;
            *(v4f*)(encodings + rowbase + k) = v;
        }
    }
}

// ---------------------------------------------------------------------------
// Kernel 3: EMA epilogue (1 block x 1024 threads, t == code index)
// ---------------------------------------------------------------------------
__device__ __forceinline__ float block_reduce_1024(float v, float* red, int t) {
#pragma unroll
    for (int off = 16; off; off >>= 1) v += __shfl_xor(v, off, 32);
    if ((t & 31) == 0) red[t >> 5] = v;
    __syncthreads();
    if (t < 32) {
        float s = red[t];
#pragma unroll
        for (int off = 16; off; off >>= 1) s += __shfl_xor(s, off, 32);
        if (t == 0) red[32] = s;
    }
    __syncthreads();
    float r = red[32];
    __syncthreads();
    return r;
}

__global__ __launch_bounds__(1024) void vq_finalize(
    const float* __restrict__ ema_cluster_size,  // [K]
    const float* __restrict__ ema_w,             // [K, D]
    const float* __restrict__ dw,                // [K, D]
    const float* __restrict__ counts,            // [K]
    const float* __restrict__ loss_part,         // [4096]
    float* __restrict__ out_loss,
    float* __restrict__ out_entropy,
    float* __restrict__ out_cluster,             // [K]
    float* __restrict__ out_ema_w,               // [K, D]
    float* __restrict__ out_embedding)           // [K, D]
{
    __shared__ float red[33];
    __shared__ float clnorm[KCODES];
    const int t = threadIdx.x;

    const float cnt = counts[t];
    const float cl  = ema_cluster_size[t] * DECAYF + ONE_MINUS_DECAYF * cnt;

    const float nsum = block_reduce_1024(cl, red, t);
    const float cln  = (cl + EPSF) / (nsum + 1024.0f * EPSF) * nsum;
    out_cluster[t] = cln;
    clnorm[t]      = cln;

    const float p   = cnt * (1.0f / 65536.0f);
    const float ent = block_reduce_1024(-p * logf(p + 1e-10f), red, t);

    float lp = loss_part[t] + loss_part[t + 1024] + loss_part[t + 2048] + loss_part[t + 3072];
    const float lsum = block_reduce_1024(lp, red, t);

    if (t == 0) {
        out_entropy[0] = ent;
        out_loss[0]    = 0.25f * (lsum * (1.0f / 16777216.0f));
    }
    __syncthreads();

    for (int i = t; i < KCODES * DDIM; i += 1024) {
        float w = ema_w[i] * DECAYF + ONE_MINUS_DECAYF * dw[i];
        out_ema_w[i]     = w;
        out_embedding[i] = w / clnorm[i >> 8];
    }
}

// ---------------------------------------------------------------------------
extern "C" void kernel_launch(void* const* d_in, const int* in_sizes, int n_in,
                              void* d_out, int out_size, void* d_ws, size_t ws_size,
                              hipStream_t stream) {
    const float* inputs           = (const float*)d_in[0];   // [64,256,32,32]
    const float* embedding        = (const float*)d_in[1];   // [1024,256]
    const float* ema_cluster_size = (const float*)d_in[2];   // [1024]
    const float* ema_w            = (const float*)d_in[3];   // [1024,256]

    float* o           = (float*)d_out;
    float* out_x       = o;                                  // 16,777,216
    float* out_loss    = o + 16777216;                       // 1
    float* out_ent     = o + 16777217;                       // 1
    float* out_enc     = o + 16777218;                       // 67,108,864
    float* out_cluster = o + 83886082;                       // 1024
    float* out_emaw    = out_cluster + 1024;                 // 262,144
    float* out_emb     = out_emaw + 262144;                  // 262,144

    char* ws = (char*)d_ws;
    v16bf* ehi      = (v16bf*)(ws);                                    // 512 KB
    v16bf* elo      = (v16bf*)(ws + 524288);                           // 512 KB
    float* enorm    = (float*)(ws + 1048576);                          // 4 KB (zeroed)
    float* dwbuf    = (float*)(ws + 1048576 + 4096);                   // 1 MB (zeroed)
    float* counts   = (float*)(ws + 1048576 + 4096 + 1048576);         // 4 KB (zeroed)
    float* losspart = (float*)(ws + 1048576 + 4096 + 1048576 + 4096);  // 16 KB (fully rewritten)

    // zero the accumulators (enorm + dw + counts) -- graph-capture safe
    (void)hipMemsetAsync(enorm, 0, 4096 + 1048576 + 4096, stream);

    vq_prepack<<<512, 32, 0, stream>>>(embedding, ehi, elo, enorm);
    vq_main<<<4096, 256, 0, stream>>>(inputs, embedding, ehi, elo, enorm,
                                      out_x, out_enc, dwbuf, counts, losspart);
    vq_finalize<<<1, 1024, 0, stream>>>(ema_cluster_size, ema_w, dwbuf, counts, losspart,
                                        out_loss, out_ent, out_cluster, out_emaw, out_emb);
}
